// ModulatedConv2d_64063732187395
// MI455X (gfx1250) — matrix operand order repaired and checked by reading, not compiled, not run on hardware
//
#include <hip/hip_runtime.h>
#include <hip/hip_bf16.h>

// ---------------------------------------------------------------------------
// ModulatedConv2d (StyleGAN2) for MI455X / gfx1250.
// B=8, CIN=COUT=128, K=3, SDIM=512, H=W=256.
//   k0: x f32 NCHW -> bf16 NHWC with 1-px zero halo (workspace)
//   k1: style linear + demod + bf16 weight pack [b][pos][cout][cin]
//   k2: implicit-GEMM conv; TDM tensor_load_to_lds tile staging +
//       v_wmma_f32_16x16x32_bf16 (wave32), non-temporal output stores.
// ---------------------------------------------------------------------------

typedef __attribute__((ext_vector_type(16))) __bf16 v16bf;
typedef __attribute__((ext_vector_type(8)))  __bf16 v8bf;
typedef __attribute__((ext_vector_type(8)))  float  v8f;
typedef __attribute__((ext_vector_type(4)))  unsigned int v4u;
typedef __attribute__((ext_vector_type(8)))  int    v8i;
typedef __attribute__((ext_vector_type(4)))  int    v4i;

#define BATCH 8
#define CIN   128
#define COUT  128
#define SDIM  512
#define HH    256
#define WW    256
#define NPOS  9
#define TILE_PIX 128
#define CCHUNK 32
#define LCOLS 130
#define HALO_W (WW + 2)      // 258
#define HALO_H (HH + 2)      // 258

// ---------------------------------------------------------------------------
// Kernel 0: x [B,CIN,H,W] f32  ->  xbf [B,258,258,CIN] bf16, zero halo.
// One block per (b, padded row). LDS transpose so both global reads (over x)
// and global writes (over cin) are contiguous.
// ---------------------------------------------------------------------------
__global__ __launch_bounds__(256)
void pack_x_kernel(const float* __restrict__ x, __bf16* __restrict__ xbf)
{
    const int bid = blockIdx.x;              // b*258 + yp
    const int b   = bid / HALO_H;
    const int yp  = bid - b * HALO_H;
    const int tid = threadIdx.x;
    __bf16* row = xbf + ((size_t)b * HALO_H + yp) * HALO_W * CIN;

    if (yp == 0 || yp == HALO_H - 1) {       // top/bottom halo rows: zero
        for (int i = tid; i < HALO_W * CIN; i += 256) row[i] = (__bf16)0.f;
        return;
    }
    const int y = yp - 1;
    if (tid < CIN) {                          // left/right halo columns: zero
        row[tid] = (__bf16)0.f;
        row[(size_t)(HALO_W - 1) * CIN + tid] = (__bf16)0.f;
    }

    __shared__ float st[256 * 33];            // [x][cin] padded vs bank conflicts
    for (int cc = 0; cc < CIN / CCHUNK; ++cc) {
        __syncthreads();
        for (int i = tid; i < CCHUNK * WW; i += 256) {     // coalesced over x
            const int ci = i >> 8;
            const int xx = i & 255;
            st[xx * 33 + ci] = __builtin_nontemporal_load(
                x + (((size_t)b * CIN + cc * CCHUNK + ci) * HH + y) * WW + xx);
        }
        __syncthreads();
        for (int i = tid; i < WW * CCHUNK; i += 256) {     // coalesced over cin
            const int xx = i >> 5;
            const int ci = i & 31;
            row[(size_t)(xx + 1) * CIN + cc * CCHUNK + ci] = (__bf16)st[xx * 33 + ci];
        }
    }
}

// ---------------------------------------------------------------------------
// Kernel 1: style linear + demodulation + bf16 weight pack [b][pos][cout][cin]
// ---------------------------------------------------------------------------
__global__ __launch_bounds__(256)
void modconv_prep_kernel(const float* __restrict__ w,        // [B, SDIM]
                         const float* __restrict__ weight,   // [COUT, CIN, 3, 3]
                         const float* __restrict__ styleW,   // [CIN, SDIM]
                         const float* __restrict__ styleb,   // [CIN]
                         __bf16* __restrict__ wmod)          // [B, 9, COUT, CIN]
{
    const int b   = blockIdx.x;
    const int tid = threadIdx.x;
    __shared__ float s_style[CIN];
    __shared__ float s_dscale[COUT];

    if (tid < CIN) {
        float acc = 0.f;
        const float* wr = w + (size_t)b * SDIM;
        const float* sr = styleW + (size_t)tid * SDIM;
        for (int s = 0; s < SDIM; ++s) acc += wr[s] * sr[s];
        s_style[tid] = acc + styleb[tid];
    }
    __syncthreads();

    if (tid < COUT) {
        float ssum = 0.f;
        for (int ci = 0; ci < CIN; ++ci) {
            const float st = s_style[ci];
            const float* wp = weight + ((size_t)tid * CIN + ci) * NPOS;
            for (int p = 0; p < NPOS; ++p) {
                float v = wp[p] * st;
                ssum += v * v;
            }
        }
        s_dscale[tid] = rsqrtf(ssum + 1e-8f);
    }
    __syncthreads();

    const int total = NPOS * COUT * CIN;
    for (int i = tid; i < total; i += 256) {
        const int pos = i / (COUT * CIN);
        const int rem = i - pos * (COUT * CIN);
        const int co  = rem >> 7;
        const int ci  = rem & 127;
        const float v = weight[((size_t)co * CIN + ci) * NPOS + pos]
                        * s_style[ci] * s_dscale[co];
        wmod[(((size_t)b * NPOS + pos) * COUT + co) * CIN + ci] = (__bf16)v;
    }
}

// ---------------------------------------------------------------------------
// Kernel 2: implicit-GEMM conv. One block = one sample x 128-pixel row
// segment.  Input tile (3 rows x 130 cols x 32 cins, bf16) staged into LDS
// by the Tensor Data Mover (tensor_load_to_lds, TENSORcnt).  8 waves; wave w
// computes couts [16w,16w+16) x 128 pixels as 8 WMMA tiles.
// ---------------------------------------------------------------------------
__global__ __launch_bounds__(256)
void modconv_wmma_kernel(const __bf16* __restrict__ xbf,   // [B,258,258,CIN]
                         const __bf16* __restrict__ wmod,  // [B,9,COUT,CIN]
                         const float*  __restrict__ bias,  // [COUT]
                         float* __restrict__ out)          // [B,COUT,H,W]
{
    const int bid  = blockIdx.x;
    const int b    = bid >> 9;
    const int tile = bid & 511;
    const int y    = tile >> 1;          // image row
    const int x0   = (tile & 1) << 7;    // 0 or 128
    const int tid  = threadIdx.x;
    const int lane = tid & 31;
    const int wave = tid >> 5;

    // LDS input tile: [3 rows][130 cols][32 cins] bf16 = 24960 B.
    __shared__ __attribute__((aligned(64))) __bf16 sh[3 * LCOLS * CCHUNK];

    v8f acc[8];
    #pragma unroll
    for (int t = 0; t < 8; ++t) acc[t] = (v8f){0.f,0.f,0.f,0.f,0.f,0.f,0.f,0.f};

    const int mrow   = lane & 15;            // A row / B column within 16
    const int aK0    = (lane & 16) ? 16 : 0; // contiguous K split (k = 16h + s)
    const int bKhalf = (lane >> 4) << 4;     // contiguous K split (k = 16h + s), same as A

    union AFrag { v16bf v; v8bf h[2]; };

    // ---- TDM descriptor constants (D# per ISA ch.8) ----------------------
    // 3D tile: X = 32 cin (contig), Y = 130 cols (stride 128), Z = 3 rows
    // (stride 258*128).  data_size = 2 bytes.  Halo => tile never OOB.
    const unsigned lds_base = (unsigned)(size_t)(&sh[0]);
    // tile start (halo coords): row y' = y, col x' = x0
    const unsigned long long tile_elem0 =
        (((unsigned long long)b * HALO_H + (unsigned)y) * HALO_W + (unsigned)x0) * CIN;

    for (int cc = 0; cc < CIN / CCHUNK; ++cc) {
        __syncthreads();                       // prior chunk's LDS reads done
        if (tid < 32) {
            const unsigned long long ga =
                (unsigned long long)(size_t)xbf + 2ull * (tile_elem0 + (unsigned)(cc * CCHUNK));
            v4u g0;
            g0.x = 1u;                                     // count=1, load D#
            g0.y = lds_base;                               // lds_addr
            g0.z = (unsigned)ga;                           // global_addr[31:0]
            g0.w = (unsigned)((ga >> 32) & 0x01FFFFFFu) | (2u << 30); // [56:32] | type=2
            v8i g1;
            g1[0] = 0x00010000;                // mask=0, data_size=1 (2B)
            g1[1] = (CCHUNK << 16);            // tensor_dim0 lo = 32
            g1[2] = (LCOLS  << 16);            // dim0 hi=0 | tensor_dim1 lo = 130
            g1[3] = (CCHUNK << 16);            // dim1 hi=0 | tile_dim0 = 32
            g1[4] = (3 << 16) | LCOLS;         // tile_dim1 = 130 | tile_dim2 = 3
            g1[5] = CIN;                       // tensor_dim0_stride = 128
            g1[6] = (int)((unsigned)(HALO_W * CIN) << 16); // dim1_stride lo16 = 33024
            g1[7] = (int)((unsigned)(HALO_W * CIN) >> 16); // dim1_stride hi
            v4i g2;
            g2[0] = 3;                         // tensor_dim2 = 3
            g2[1] = 0; g2[2] = 0; g2[3] = 0;   // no dim3 / no iterate
            v4i g3; g3[0] = 0; g3[1] = 0; g3[2] = 0; g3[3] = 0;
            v8i g4;                             // extra group (6-arg builtin): unused
            g4[0] = 0; g4[1] = 0; g4[2] = 0; g4[3] = 0;
            g4[4] = 0; g4[5] = 0; g4[6] = 0; g4[7] = 0;
            __builtin_amdgcn_tensor_load_to_lds(g0, g1, g2, g3, g4, 0);
            __builtin_amdgcn_s_wait_tensorcnt(0);
        }
        __syncthreads();                       // tile visible to all waves

        #pragma unroll
        for (int pos = 0; pos < NPOS; ++pos) {
            const int ky = pos / 3, kx = pos - ky * 3;

            // A fragment: wmod[b][pos][wave*16 + mrow][cc*32 + k]
            const __bf16* wp = wmod
                + (((size_t)b * NPOS + pos) * COUT + (wave * 16 + mrow)) * CIN
                + cc * CCHUNK + aK0;
            AFrag a;
            a.h[0] = *(const v8bf*)(wp);
            a.h[1] = *(const v8bf*)(wp + 8);

            if (pos + 1 < NPOS)
                __builtin_prefetch(wp + (size_t)COUT * CIN, 0, 2);

            #pragma unroll
            for (int nt = 0; nt < 8; ++nt) {
                const int col = nt * 16 + mrow + kx;
                const v16bf bf = *(const v16bf*)
                    (sh + (ky * LCOLS + col) * CCHUNK + bKhalf);
                acc[nt] = __builtin_amdgcn_wmma_f32_16x16x32_bf16(
                    false, a.v, false, bf, (short)0, acc[nt], false, false);
            }
        }
    }

    // ---- epilogue: C/D layout -> out[b][cout][y][x], non-temporal --------
    const int mhalf = lane >> 4;
    const int n     = lane & 15;
    #pragma unroll
    for (int nt = 0; nt < 8; ++nt) {
        const int xx = x0 + nt * 16 + n;
        #pragma unroll
        for (int r = 0; r < 8; ++r) {
            const int co = wave * 16 + r + 8 * mhalf;
            __builtin_nontemporal_store(
                acc[nt][r] + bias[co],
                out + (((size_t)b * COUT + co) * HH + y) * WW + xx);
        }
    }
}

// ---------------------------------------------------------------------------
extern "C" void kernel_launch(void* const* d_in, const int* in_sizes, int n_in,
                              void* d_out, int out_size, void* d_ws, size_t ws_size,
                              hipStream_t stream) {
    const float* x       = (const float*)d_in[0];  // [8,128,256,256]
    const float* w       = (const float*)d_in[1];  // [8,512]
    const float* weight  = (const float*)d_in[2];  // [128,128,3,3]
    const float* styleW  = (const float*)d_in[3];  // [128,512]
    const float* styleb  = (const float*)d_in[4];  // [128]
    const float* bias    = (const float*)d_in[5];  // [128]
    float*       out     = (float*)d_out;          // [8,128,256,256]

    // workspace layout: wmod bf16 (2.25 MB) @ 0, xbf bf16 (~130 MB) @ 4 MB
    __bf16* wmod = (__bf16*)d_ws;
    __bf16* xbf  = (__bf16*)((char*)d_ws + (4u << 20));

    pack_x_kernel<<<BATCH * HALO_H, 256, 0, stream>>>(x, xbf);
    modconv_prep_kernel<<<BATCH, 256, 0, stream>>>(w, weight, styleW, styleb, wmod);

    const int nblocks = BATCH * (HH * WW / TILE_PIX);   // 4096
    modconv_wmma_kernel<<<nblocks, 256, 0, stream>>>(xbf, wmod, bias, out);
}
